// DNRI_MLP_Decoder_49357764166098
// MI455X (gfx1250) — matrix operand order repaired and checked by reading, not compiled, not run on hardware
//
#include <hip/hip_runtime.h>
#include <hip/hip_bf16.h>

// ---------------------------------------------------------------------------
// DNRI MLP decoder for MI455X (gfx1250), wave32 + v_wmma_f32_16x16x32_f16.
//
// Pipeline:
//   k0: convert all weights f32 -> f16 into workspace (fc1 padded K 48->64).
//   k1: per (b, receiver r) wave: 49 incoming edges (senders s != r),
//       edge MLP (32->32->32, relu) per type via WMMA, scale by edges[b,e,k],
//       reduce over edges -> agg[b,r,32]  (no atomics, no [B,E,32] buffer).
//   k2: per 16-row node tile: aug(48) ->fc1(256) relu ->fc2(256) relu ->mu(16)
//       via WMMA; writes pred and mu directly into d_out.
//
// Tile-build LDS writes are packed as v8h (ds_store_b128); global feature
// reads are float4 (global_load_b128).
// ---------------------------------------------------------------------------

#define NVAR 50
#define INSZ 16
#define MHID 32
#define NHID 256
#define NEDGE 2450
#define NBATCH 512
#define NROWS (NBATCH * NVAR)   // 25600 node rows
#define NTILES (NROWS / 16)     // 1600 (exact)
#define NPAIRS (NBATCH * NVAR)  // 25600 (exact multiple of 4)

typedef _Float16 v16h __attribute__((ext_vector_type(16)));
typedef _Float16 v8h  __attribute__((ext_vector_type(8)));
typedef float    v8f  __attribute__((ext_vector_type(8)));

__device__ __forceinline__ v8f wmma_f16(v16h a, v16h b, v8f c) {
  // (neg_a, A, neg_b, B, c_mod, C, reuse_a, reuse_b)
  return __builtin_amdgcn_wmma_f32_16x16x32_f16(false, a, false, b,
                                                (short)0, c, false, false);
}

// Read a 16x32 f16 tile (row-major in LDS, rowp = base + (lane&15)*ld + chunk)
// into the ISA A-layout: lanes 0-15 hold K {0..7,16..23}, lanes 16-31 hold
// K {8..15,24..31}.  Two 16-B LDS loads.
__device__ __forceinline__ v16h read_a16(const _Float16* rowp, int lane) {
  int kg = (lane < 16) ? 0 : 8;
  v8h lo = *(const v8h*)(rowp + kg);
  v8h hi = *(const v8h*)(rowp + 16 + kg);
  return __builtin_shufflevector(lo, hi, 0, 1, 2, 3, 4, 5, 6, 7,
                                 8, 9, 10, 11, 12, 13, 14, 15);
}

__device__ __forceinline__ v16h load_b16(const _Float16* p, int ld, int lane) {
  int n = lane & 15, kg = (lane < 16) ? 0 : 8;
  v16h r;
#pragma unroll
  for (int s = 0; s < 8; ++s) {
    r[s] = p[(size_t)(kg + s) * ld + n];
    r[8 + s] = p[(size_t)(16 + kg + s) * ld + n];
  }
  return r;
}

// pack 8 consecutive f32 (two float4) into v8h
__device__ __forceinline__ v8h pack8(float4 a, float4 b) {
  v8h p;
  p[0] = (_Float16)a.x; p[1] = (_Float16)a.y;
  p[2] = (_Float16)a.z; p[3] = (_Float16)a.w;
  p[4] = (_Float16)b.x; p[5] = (_Float16)b.y;
  p[6] = (_Float16)b.z; p[7] = (_Float16)b.w;
  return p;
}

// ---------------------------------------------------------------------------
// Kernel 0: weight conversion f32 -> f16 (fc1 zero-padded K: 48 -> 64).
// ---------------------------------------------------------------------------
__global__ __launch_bounds__(256) void convert_weights_kernel(
    const float* __restrict__ w1, const float* __restrict__ w2,
    const float* __restrict__ fc1, const float* __restrict__ fc2,
    const float* __restrict__ mu,
    _Float16* __restrict__ w1h, _Float16* __restrict__ w2h,
    _Float16* __restrict__ fc1h, _Float16* __restrict__ fc2h,
    _Float16* __restrict__ muh) {
  int i = blockIdx.x * blockDim.x + threadIdx.x;
  if (i < 4096) {
    w1h[i] = (_Float16)w1[i];
    w2h[i] = (_Float16)w2[i];
    muh[i] = (_Float16)mu[i];
  }
  if (i < 65536) fc2h[i] = (_Float16)fc2[i];
  if (i < 16384) {  // padded [64][256]
    int k = i >> 8, n = i & 255;
    fc1h[i] = (k < 48) ? (_Float16)fc1[k * 256 + n] : (_Float16)0.f;
  }
}

// ---------------------------------------------------------------------------
// Kernel 1: edge message MLPs + per-receiver aggregation.
// One wave per (b, r). 4 waves per block. Grid = NPAIRS/4 (exact).
// ---------------------------------------------------------------------------
__global__ __launch_bounds__(128) void edge_msg_kernel(
    const float* __restrict__ inputs, const float* __restrict__ edges,
    const _Float16* __restrict__ w1h, const float* __restrict__ b1,
    const _Float16* __restrict__ w2h, const float* __restrict__ b2,
    float* __restrict__ agg) {
  __shared__ _Float16 pre_s[4][16][32];
  __shared__ _Float16 h_s[4][16][32];

  const int wave = threadIdx.x >> 5;
  const int lane = threadIdx.x & 31;
  const int pair = blockIdx.x * 4 + wave;
  const int b = pair / NVAR;
  const int r = pair % NVAR;
  const int col = lane & 15;
  const int half = lane >> 4;

  _Float16 (*pre)[32] = pre_s[wave];
  _Float16 (*hs)[32] = h_s[wave];

  // receiver features: wave-uniform 64B, packed once into registers
  const float4* rf4 = (const float4*)(inputs + ((size_t)b * NVAR + r) * INSZ);
  v8h recv0 = pack8(rf4[0], rf4[1]);
  v8h recv1 = pack8(rf4[2], rf4[3]);

  float acc0 = 0.f, acc1 = 0.f;  // output features col and 16+col

  for (int t = 0; t < 4; ++t) {
    // ---- build A tile (rows = 16 sender slots) + edge coefficients ----
    float cf[4] = {0.f, 0.f, 0.f, 0.f};
    if (lane < 16) {
      int q = t * 16 + lane;  // sender ordinal, 0..48 valid
      bool valid = q < 49;
      int s = q + (q >= r ? 1 : 0);
      _Float16* row = pre[lane];
      if (valid) {
        const float4* sf4 =
            (const float4*)(inputs + ((size_t)b * NVAR + s) * INSZ);
        *(v8h*)(row + 0) = recv0;                     // receiver (cols 0..15)
        *(v8h*)(row + 8) = recv1;
        *(v8h*)(row + 16) = pack8(sf4[0], sf4[1]);    // sender (cols 16..31)
        *(v8h*)(row + 24) = pack8(sf4[2], sf4[3]);
        int e = s * 49 + (r < s ? r : r - 1);
        const float* ep = edges + ((size_t)b * NEDGE + e) * 4;
        cf[1] = ep[1];
        cf[2] = ep[2];
        cf[3] = ep[3];
      } else {
        v8h z = {};
        *(v8h*)(row + 0) = z;
        *(v8h*)(row + 8) = z;
        *(v8h*)(row + 16) = z;
        *(v8h*)(row + 24) = z;
      }
    }
    __syncthreads();
    v16h a0 = read_a16(&pre[col][0], lane);

#pragma unroll
    for (int k = 1; k < 4; ++k) {
      // ---- layer 1: 16x32 @ 32x32 (two N-halves) + relu -> LDS ----
#pragma unroll
      for (int hn = 0; hn < 2; ++hn) {
        v8f c;
        float bias = b1[k * 32 + hn * 16 + col];
#pragma unroll
        for (int v = 0; v < 8; ++v) c[v] = bias;
        v16h bv = load_b16(w1h + (size_t)k * 32 * 32 + hn * 16, 32, lane);
        c = wmma_f16(a0, bv, c);
        int rb = half * 8;
#pragma unroll
        for (int v = 0; v < 8; ++v)
          hs[rb + v][hn * 16 + col] = (_Float16)fmaxf(c[v], 0.f);
      }
      __syncthreads();
      v16h a1 = read_a16(&hs[col][0], lane);
      __syncthreads();  // everyone has read hs before next k overwrites it

      // ---- layer 2 + relu + edge-coefficient scale + row reduction ----
#pragma unroll
      for (int hn = 0; hn < 2; ++hn) {
        v8f c;
        float bias = b2[k * 32 + hn * 16 + col];
#pragma unroll
        for (int v = 0; v < 8; ++v) c[v] = bias;
        v16h bv = load_b16(w2h + (size_t)k * 32 * 32 + hn * 16, 32, lane);
        c = wmma_f16(a1, bv, c);
#pragma unroll
        for (int v = 0; v < 8; ++v) {
          float x = fmaxf(c[v], 0.f);
          float cm = __shfl(cf[k], v + half * 8, 32);  // coef of row M
          x *= cm;
          if (hn == 0) acc0 += x; else acc1 += x;
        }
      }
    }
    __syncthreads();  // before next tile overwrites pre
  }

  // combine the two row-halves: lane l and lane l+16 hold complementary rows
  acc0 += __shfl_xor(acc0, 16, 32);
  acc1 += __shfl_xor(acc1, 16, 32);
  float out = (lane < 16) ? acc0 : acc1;  // feature index == lane
  agg[((size_t)b * NVAR + r) * 32 + lane] = out;
}

// ---------------------------------------------------------------------------
// Kernel 2: node MLP. One wave per 16-row tile, 4 waves/block.
// Grid = NTILES/4 = 400 (exact).
// ---------------------------------------------------------------------------
__global__ __launch_bounds__(128) void node_mlp_kernel(
    const float* __restrict__ inputs, const float* __restrict__ agg,
    const _Float16* __restrict__ fc1h, const float* __restrict__ fc1b,
    const _Float16* __restrict__ fc2h, const float* __restrict__ fc2b,
    const _Float16* __restrict__ muh, const float* __restrict__ mub,
    float* __restrict__ mu_out, float* __restrict__ pred_out) {
  __shared__ _Float16 aug_s[4][16][64];
  __shared__ _Float16 h_s[4][16][256];

  const int wave = threadIdx.x >> 5;
  const int lane = threadIdx.x & 31;
  const int tile = blockIdx.x * 4 + wave;
  const int col = lane & 15;
  const int rb = (lane >> 4) * 8;

  _Float16 (*aug)[64] = aug_s[wave];
  _Float16 (*hs)[256] = h_s[wave];

  // ---- build aug = [inputs(16) | agg(32) | zeros(16)] in f16, packed ----
  if (lane < 16) {
    int row = tile * 16 + lane;  // < 25600 guaranteed
    const float4* ip4 = (const float4*)(inputs + (size_t)row * INSZ);
    const float4* ap4 = (const float4*)(agg + (size_t)row * 32);
    _Float16* dst = aug[lane];
    *(v8h*)(dst + 0) = pack8(ip4[0], ip4[1]);
    *(v8h*)(dst + 8) = pack8(ip4[2], ip4[3]);
    *(v8h*)(dst + 16) = pack8(ap4[0], ap4[1]);
    *(v8h*)(dst + 24) = pack8(ap4[2], ap4[3]);
    *(v8h*)(dst + 32) = pack8(ap4[4], ap4[5]);
    *(v8h*)(dst + 40) = pack8(ap4[6], ap4[7]);
    v8h z = {};
    *(v8h*)(dst + 48) = z;
    *(v8h*)(dst + 56) = z;
  }
  __syncthreads();
  v16h a0 = read_a16(&aug[col][0], lane);
  v16h a1 = read_a16(&aug[col][32], lane);

  // ---- fc1: 16x64 @ 64x256 (+bias, relu) -> hs ----
  for (int hn = 0; hn < 16; ++hn) {
    v8f c;
    float bias = fc1b[hn * 16 + col];
#pragma unroll
    for (int v = 0; v < 8; ++v) c[v] = bias;
    v16h b0 = load_b16(fc1h + hn * 16, 256, lane);
    c = wmma_f16(a0, b0, c);
    v16h b1 = load_b16(fc1h + (size_t)32 * 256 + hn * 16, 256, lane);
    c = wmma_f16(a1, b1, c);
#pragma unroll
    for (int v = 0; v < 8; ++v)
      hs[rb + v][hn * 16 + col] = (_Float16)fmaxf(c[v], 0.f);
  }
  __syncthreads();
  v16h ah[8];
#pragma unroll
  for (int ch = 0; ch < 8; ++ch) ah[ch] = read_a16(&hs[col][ch * 32], lane);
  __syncthreads();  // hs will be overwritten with h2

  // ---- fc2: 16x256 @ 256x256 (+bias, relu) -> pred (global) + hs ----
  for (int hn = 0; hn < 16; ++hn) {
    v8f c;
    float bias = fc2b[hn * 16 + col];
#pragma unroll
    for (int v = 0; v < 8; ++v) c[v] = bias;
#pragma unroll
    for (int ch = 0; ch < 8; ++ch) {
      v16h bv = load_b16(fc2h + (size_t)ch * 32 * 256 + hn * 16, 256, lane);
      c = wmma_f16(ah[ch], bv, c);
    }
#pragma unroll
    for (int v = 0; v < 8; ++v) {
      float x = fmaxf(c[v], 0.f);
      hs[rb + v][hn * 16 + col] = (_Float16)x;
      int grow = tile * 16 + rb + v;
      pred_out[(size_t)grow * 256 + hn * 16 + col] = x;
    }
  }
  __syncthreads();
#pragma unroll
  for (int ch = 0; ch < 8; ++ch) ah[ch] = read_a16(&hs[col][ch * 32], lane);

  // ---- mu: 16x256 @ 256x16 (+bias, no relu) ----
  {
    v8f c;
    float bias = mub[col];
#pragma unroll
    for (int v = 0; v < 8; ++v) c[v] = bias;
#pragma unroll
    for (int ch = 0; ch < 8; ++ch) {
      v16h bv = load_b16(muh + (size_t)ch * 32 * 16, 16, lane);
      c = wmma_f16(ah[ch], bv, c);
    }
#pragma unroll
    for (int v = 0; v < 8; ++v)
      mu_out[(size_t)(tile * 16 + rb + v) * 16 + col] = c[v];
  }
}

// ---------------------------------------------------------------------------
extern "C" void kernel_launch(void* const* d_in, const int* in_sizes, int n_in,
                              void* d_out, int out_size, void* d_ws,
                              size_t ws_size, hipStream_t stream) {
  const float* inputs = (const float*)d_in[0];
  const float* edges = (const float*)d_in[1];
  const float* w1 = (const float*)d_in[2];
  const float* b1 = (const float*)d_in[3];
  const float* w2 = (const float*)d_in[4];
  const float* b2 = (const float*)d_in[5];
  const float* fc1w = (const float*)d_in[6];
  const float* fc1b = (const float*)d_in[7];
  const float* fc2w = (const float*)d_in[8];
  const float* fc2b = (const float*)d_in[9];
  const float* muw = (const float*)d_in[10];
  const float* mub = (const float*)d_in[11];

  float* mu_out = (float*)d_out;
  float* pred_out = (float*)d_out + (size_t)NBATCH * NVAR * INSZ;

  // workspace layout (needs ~3.5 MB)
  char* ws = (char*)d_ws;
  float* agg = (float*)ws;                 // 512*50*32 f32 = 3,276,800 B
  size_t off = (size_t)NBATCH * NVAR * 32 * sizeof(float);
  _Float16* w1h = (_Float16*)(ws + off);   off += 4096 * 2;   // [4][32][32]
  _Float16* w2h = (_Float16*)(ws + off);   off += 4096 * 2;   // [4][32][32]
  _Float16* fc1h = (_Float16*)(ws + off);  off += 16384 * 2;  // [64][256] padded
  _Float16* fc2h = (_Float16*)(ws + off);  off += 65536 * 2;  // [256][256]
  _Float16* muh = (_Float16*)(ws + off);   off += 4096 * 2;   // [256][16]

  convert_weights_kernel<<<256, 256, 0, stream>>>(w1, w2, fc1w, fc2w, muw, w1h,
                                                  w2h, fc1h, fc2h, muh);
  edge_msg_kernel<<<NPAIRS / 4, 128, 0, stream>>>(inputs, edges, w1h, b1, w2h,
                                                  b2, agg);
  node_mlp_kernel<<<NTILES / 4, 128, 0, stream>>>(
      inputs, agg, fc1h, fc1b, fc2h, fc2b, muh, mub, mu_out, pred_out);
}
